// tinystoriesgru_3453153706518
// MI455X (gfx1250) — compile-verified, matrix operand order checked by reading, not run on hardware
//
#include <hip/hip_runtime.h>

#define Bsz 2048
#define Tsz 256
#define Esz 124
#define Vsz 65
#define KC  256   // padded K for [x,m] (2E=248 -> 256)
#define KH  128   // padded K for h (124 -> 128)
#define NG  384   // gate-padded 3*128
#define KCT 8     // K steps of 32 for comb
#define KHT 4     // K steps of 32 for h
#define NVT 5     // head N tiles (65 -> 80)
#define NTHREADS 256

typedef __attribute__((ext_vector_type(16))) __bf16 v16bf;
typedef __attribute__((ext_vector_type(8)))  __bf16 v8bf;
typedef __attribute__((ext_vector_type(8)))  float  v8f;

static __device__ __forceinline__ unsigned short f2bf(float f) {
  union { float f; unsigned u; } v; v.f = f;
  unsigned r = v.u + 0x7FFFu + ((v.u >> 16) & 1u);   // round-to-nearest-even
  return (unsigned short)(r >> 16);
}

static __device__ __forceinline__ float sigmoidf(float x) { return 1.0f / (1.0f + __expf(-x)); }

// A-matrix fragment (16x32 bf16, M x K) from row-major [16][stride] LDS tile.
// lanes 0-15: M=lane, VGPR0-3 = K 0..7, VGPR4-7 = K16..23; lanes 16-31: K+8 / K+24.
static __device__ __forceinline__ v16bf loadA(const unsigned short* base, int stride,
                                              int kk, int m, int hl) {
  const unsigned short* p = base + m * stride + kk * 32 + hl * 8;
  v8bf lo = *(const v8bf*)p;
  v8bf hi = *(const v8bf*)(p + 16);
  return __builtin_shufflevector(lo, hi, 0,1,2,3,4,5,6,7,8,9,10,11,12,13,14,15);
}

// B-matrix fragment (32x16 bf16, K x N), pre-swizzled so lane reads 32B contiguous.
static __device__ __forceinline__ v16bf loadB(const unsigned short* frag, int lane) {
  const unsigned short* p = frag + lane * 16;
  v8bf lo = *(const v8bf*)p;
  v8bf hi = *(const v8bf*)(p + 8);
  return __builtin_shufflevector(lo, hi, 0,1,2,3,4,5,6,7,8,9,10,11,12,13,14,15);
}

static __device__ __forceinline__ v8f wmma_bf16(v16bf a, v16bf b, v8f c) {
  return __builtin_amdgcn_wmma_f32_16x16x32_bf16(false, a, false, b, (short)0, c, false, false);
}

// C/D 16x16 f32: lane n = l&15, VGPR v -> row v + 8*(l>>4)
static __device__ __forceinline__ void storeC(v8f acc, float* base, int stride,
                                              int colbase, int lane) {
  int n = lane & 15, hl = lane >> 4;
  #pragma unroll
  for (int v = 0; v < 8; ++v) base[(v + 8 * hl) * stride + colbase + n] = acc[v];
}

// Repack f32 weight matrix [N][K] into bf16 WMMA-B fragments (B(k,n) = W[n][k]).
// gated=1 remaps 3 gate sections of width 124 onto 128-aligned sections.
__global__ void pack_frags(const float* __restrict__ src, unsigned short* __restrict__ dst,
                           int N, int K, int ksteps, int gated) {
  int f = blockIdx.x;
  int nt = f / ksteps, kk = f - nt * ksteps;
  int tid = threadIdx.x;           // 0..511
  int l = tid >> 4, j = tid & 15;  // lane, element
  int k_local = ((j >> 3) * 16) + ((l >> 4) * 8) + (j & 7);
  int n_local = l & 15;
  int k  = kk * 32 + k_local;
  int np = nt * 16 + n_local;
  int on; bool valid;
  if (gated) { on = (np >> 7) * Esz + (np & 127); valid = (np & 127) < Esz; }
  else       { on = np;                            valid = np < N; }
  float v = (valid && k < K) ? src[(size_t)on * K + k] : 0.0f;
  dst[(size_t)f * 512 + tid] = f2bf(v);
}

__global__ __launch_bounds__(NTHREADS, 1)
void gru_scan(const int* __restrict__ idx, const float* __restrict__ tok,
              const float* __restrict__ pos,
              const float* __restrict__ b_ih, const float* __restrict__ b_hh,
              const float* __restrict__ w_p, const float* __restrict__ b_p,
              const float* __restrict__ b_m, const float* __restrict__ ln_g,
              const float* __restrict__ ln_b, const float* __restrict__ b_head,
              const unsigned short* __restrict__ fIH, const unsigned short* __restrict__ fHH,
              const unsigned short* __restrict__ fM, const unsigned short* __restrict__ fHead,
              float* __restrict__ out)
{
  __shared__ __align__(16) unsigned short combA[16][KC]; // bf16 [x_t | m], K-padded
  __shared__ __align__(16) unsigned short Hb[16][KH];    // bf16 h
  __shared__ __align__(16) unsigned short Yb[16][KH];    // bf16 LN(h)
  __shared__ float GIH[16][NG];     // gi (+ gh for r,z), gate-padded columns
  __shared__ float GHN[16][128];    // gh n-gate part (kept separate for r*h_n)
  __shared__ float CANDs[16][128];
  __shared__ float bihP[NG], bhhP[NG];
  __shared__ float bmS[Esz], lngS[Esz], lnbS[Esz], wpS[Esz], bheadS[80];

  const int tid  = threadIdx.x;
  const int wave = tid >> 5, lane = tid & 31;
  const int m_l  = lane & 15, hl = lane >> 4;
  const int row  = tid >> 4, sub = tid & 15;   // 16 threads per batch row
  const int b0   = blockIdx.x * 16;

  for (int i = tid; i < 16 * KC; i += NTHREADS) (&combA[0][0])[i] = 0;
  for (int i = tid; i < 16 * KH; i += NTHREADS) { (&Hb[0][0])[i] = 0; (&Yb[0][0])[i] = 0; }
  for (int i = tid; i < NG; i += NTHREADS) {
    int g = i >> 7, c = i & 127;
    bihP[i] = (c < Esz) ? b_ih[g * Esz + c] : 0.f;
    bhhP[i] = (c < Esz) ? b_hh[g * Esz + c] : 0.f;
  }
  for (int i = tid; i < Esz; i += NTHREADS) {
    bmS[i] = b_m[i]; lngS[i] = ln_g[i]; lnbS[i] = ln_b[i]; wpS[i] = w_p[i];
  }
  for (int i = tid; i < 80; i += NTHREADS) bheadS[i] = (i < Vsz) ? b_head[i] : 0.f;
  const float bp = b_p[0];

  // Per-thread master state: this thread owns (row, c = sub + 16*j) forever.
  float hreg[8], mreg[8];
  #pragma unroll
  for (int j = 0; j < 8; ++j) { hreg[j] = 0.f; mreg[j] = 0.f; }
  float pgate = 0.f;   // memory-gate p from previous step

  __syncthreads();

  for (int t = 0; t < Tsz; ++t) {
    // ---- phase 0: deferred m update (uses CAND(t-1)) + embed x_t ----
    if (t > 0) {
      #pragma unroll
      for (int j = 0; j < 8; ++j) {
        int c = sub + 16 * j;
        if (c < Esz) {
          float cd = tanhf(CANDs[row][c] + bmS[c]);
          float mv = (1.f - pgate) * mreg[j] + pgate * cd;
          mreg[j] = mv;
          combA[row][Esz + c] = f2bf(mv);
        }
      }
    }
    {
      int token = idx[(size_t)(b0 + row) * Tsz + t];
      const float* te = tok + (size_t)token * Esz;
      const float* pe = pos + (size_t)t * Esz;
      for (int c = sub; c < Esz; c += 16) combA[row][c] = f2bf(te[c] + pe[c]);
    }
    __syncthreads();

    // ---- GEMM1: gi (24 N-tiles x 8 K) with gh fused for r,z; separate for n ----
    {
      v16bf aC[KCT], aH[KHT];
      #pragma unroll
      for (int kk = 0; kk < KCT; ++kk) aC[kk] = loadA(&combA[0][0], KC, kk, m_l, hl);
      #pragma unroll
      for (int kk = 0; kk < KHT; ++kk) aH[kk] = loadA(&Hb[0][0], KH, kk, m_l, hl);
      #pragma unroll
      for (int ti = 0; ti < 3; ++ti) {
        int nt = wave * 3 + ti;
        v8f acc = {};
        const unsigned short* fb = fIH + (size_t)(nt * KCT) * 512;
        #pragma unroll
        for (int kk = 0; kk < KCT; ++kk) acc = wmma_bf16(aC[kk], loadB(fb + kk * 512, lane), acc);
        const unsigned short* fh = fHH + (size_t)(nt * KHT) * 512;
        if (nt < 16) {   // r,z tiles: accumulate gh into same tile
          #pragma unroll
          for (int kk = 0; kk < KHT; ++kk) acc = wmma_bf16(aH[kk], loadB(fh + kk * 512, lane), acc);
          storeC(acc, &GIH[0][0], NG, nt * 16, lane);
        } else {         // n tiles: keep gi and gh separate (need r * h_n)
          storeC(acc, &GIH[0][0], NG, nt * 16, lane);
          v8f acc2 = {};
          #pragma unroll
          for (int kk = 0; kk < KHT; ++kk) acc2 = wmma_bf16(aH[kk], loadB(fh + kk * 512, lane), acc2);
          storeC(acc2, &GHN[0][0], 128, (nt - 16) * 16, lane);
        }
      }
    }
    __syncthreads();

    // ---- phase A: gates, h update, in-wave LN + p reductions, LN output ----
    {
      float s = 0.f, q = 0.f, pd = 0.f;
      #pragma unroll
      for (int j = 0; j < 8; ++j) {
        int c = sub + 16 * j;
        if (c < Esz) {
          float rg  = sigmoidf(GIH[row][c]       + bihP[c]       + bhhP[c]);
          float zg  = sigmoidf(GIH[row][128 + c] + bihP[128 + c] + bhhP[128 + c]);
          float inn = GIH[row][256 + c] + bihP[256 + c];
          float hnn = GHN[row][c]       + bhhP[256 + c];
          float ng  = tanhf(inn + rg * hnn);
          float hv  = (1.f - zg) * ng + zg * hreg[j];
          hreg[j] = hv;
          Hb[row][c] = f2bf(hv);
          s += hv; q += hv * hv; pd += hv * wpS[c];
        }
      }
      // butterfly over the 16 lanes of this row (stays within a wave32 half)
      #pragma unroll
      for (int m = 1; m < 16; m <<= 1) {
        s  += __shfl_xor(s,  m, 32);
        q  += __shfl_xor(q,  m, 32);
        pd += __shfl_xor(pd, m, 32);
      }
      float mu  = s * (1.0f / Esz);
      float var = q * (1.0f / Esz) - mu * mu;
      float rs  = rsqrtf(var + 1e-5f);
      pgate     = sigmoidf(pd + bp);
      #pragma unroll
      for (int j = 0; j < 8; ++j) {
        int c = sub + 16 * j;
        if (c < Esz)
          Yb[row][c] = f2bf((hreg[j] - mu) * rs * lngS[c] + lnbS[c]);
      }
    }
    __syncthreads();

    // ---- GEMM2 (cand, 8 tiles) + GEMM3 (head, 5 tiles -> direct global store) ----
    {
      v16bf aH2[KHT], aY[KHT];
      #pragma unroll
      for (int kk = 0; kk < KHT; ++kk) {
        aH2[kk] = loadA(&Hb[0][0], KH, kk, m_l, hl);
        aY[kk]  = loadA(&Yb[0][0], KH, kk, m_l, hl);
      }
      {
        int nt = wave;
        v8f acc = {};
        const unsigned short* fm = fM + (size_t)(nt * KHT) * 512;
        #pragma unroll
        for (int kk = 0; kk < KHT; ++kk) acc = wmma_bf16(aH2[kk], loadB(fm + kk * 512, lane), acc);
        storeC(acc, &CANDs[0][0], 128, nt * 16, lane);
      }
      if (wave < NVT) {
        int nt = wave;
        v8f acc = {};
        const unsigned short* fv = fHead + (size_t)(nt * KHT) * 512;
        #pragma unroll
        for (int kk = 0; kk < KHT; ++kk) acc = wmma_bf16(aY[kk], loadB(fv + kk * 512, lane), acc);
        int n = nt * 16 + m_l;
        if (n < Vsz) {
          float bh = bheadS[n];
          #pragma unroll
          for (int v = 0; v < 8; ++v) {
            int mr = v + 8 * hl;
            out[((size_t)(b0 + mr) * Tsz + t) * Vsz + n] = acc[v] + bh;
          }
        }
      }
    }
    __syncthreads();
  }
}

extern "C" void kernel_launch(void* const* d_in, const int* in_sizes, int n_in,
                              void* d_out, int out_size, void* d_ws, size_t ws_size,
                              hipStream_t stream) {
  (void)in_sizes; (void)n_in; (void)out_size; (void)ws_size;
  const int*   idx    = (const int*)  d_in[0];
  const float* tok    = (const float*)d_in[1];
  const float* pos    = (const float*)d_in[2];
  const float* w_ih   = (const float*)d_in[3];
  const float* w_hh   = (const float*)d_in[4];
  const float* b_ih   = (const float*)d_in[5];
  const float* b_hh   = (const float*)d_in[6];
  const float* w_p    = (const float*)d_in[7];
  const float* b_p    = (const float*)d_in[8];
  const float* w_m    = (const float*)d_in[9];
  const float* b_m    = (const float*)d_in[10];
  const float* ln_g   = (const float*)d_in[11];
  const float* ln_b   = (const float*)d_in[12];
  const float* w_head = (const float*)d_in[13];
  const float* b_head = (const float*)d_in[14];
  float* out = (float*)d_out;

  unsigned short* ws    = (unsigned short*)d_ws;
  unsigned short* fIH   = ws;                    // 24*8 frags * 512
  unsigned short* fHH   = fIH + 24 * 8 * 512;    // 24*4
  unsigned short* fM    = fHH + 24 * 4 * 512;    //  8*4
  unsigned short* fHead = fM  +  8 * 4 * 512;    //  5*4  (total ~340 KB)

  pack_frags<<<24 * 8, 512, 0, stream>>>(w_ih,   fIH,   372, 248, 8, 1);
  pack_frags<<<24 * 4, 512, 0, stream>>>(w_hh,   fHH,   372, 124, 4, 1);
  pack_frags<<< 8 * 4, 512, 0, stream>>>(w_m,    fM,    124, 124, 4, 0);
  pack_frags<<< 5 * 4, 512, 0, stream>>>(w_head, fHead,  65, 124, 4, 0);
  gru_scan<<<Bsz / 16, NTHREADS, 0, stream>>>(idx, tok, pos, b_ih, b_hh, w_p, b_p,
                                              b_m, ln_g, ln_b, b_head,
                                              fIH, fHH, fM, fHead, out);
}
